// AdapterTemporalGNN_30674656428556
// MI455X (gfx1250) — matrix operand order repaired and checked by reading, not compiled, not run on hardware
//
#include <hip/hip_runtime.h>
#include <cstddef>

// ==========================================================================
// MI455X (gfx1250, wave32) implementation of AdapterTemporalGNN forward.
// Dense matmuls -> v_wmma_f32_16x16x32_bf16 (fp32 -> bf16 RNE operands, f32
// accumulate), double-buffered LDS B tiles shared by 8 waves/block, b128 A
// loads. Irregular GNN phases (segment mean / softmax / message scatter) ->
// flat kernels with f32 atomics.
// ==========================================================================

typedef __attribute__((ext_vector_type(16))) __bf16       v16bf;
typedef __attribute__((ext_vector_type(8)))  float        v8f;
typedef __attribute__((ext_vector_type(8)))  unsigned int v8u;

// fp32 -> bf16 bits, round-to-nearest-even
__device__ __forceinline__ unsigned short f2bf(float f) {
  unsigned u = __float_as_uint(f);
  u += 0x7FFFu + ((u >> 16) & 1u);
  return (unsigned short)(u >> 16);
}
__device__ __forceinline__ unsigned pack2(float a, float b) {
  return (unsigned)f2bf(a) | ((unsigned)f2bf(b) << 16);
}

__device__ __forceinline__ void atomicMaxF(float* addr, float val) {
  int* ai = (int*)addr;
  int old = *ai;
  while (__int_as_float(old) < val) {
    int assumed = old;
    old = atomicCAS(ai, assumed, __float_as_int(val));
    if (old == assumed) break;
  }
}

// --------------------------------------------------------------------------
// GEMM: C[rows x M] = epilogue(A[rows x K] @ W[K x M] + bias)
// Caller guarantees: rows%16==0, M%16==0, K%32==0, lda/ldw/ldc%4==0.
// Block = 256 threads = 8 waves sharing one column-tile; wave w computes
// row-tile (blockRow*8 + w). B tiles staged to LDS as bf16 in column-major
// fragment order (double-buffered, 64 K per barrier pair when K%64==0);
// each lane reads its 16-element B fragment as one aligned 32B LDS vector.
// Fragment layouts used here (k contiguous per lane half, not the ISA table order):
//   A (16x32): lane&15 = row; lanes 0-15 K 0..15, lanes 16-31 K 16..31
//   B (32x16): lane&15 = col; lanes 0-15 K 0..15, lanes 16-31 K 16..31
//   C/D     : acc[r] -> row (lane>>4)*8 + r, col lane&15
// ACT: 0=none 1=relu 2=sigmoid.  RESID: C = resid + scale*(acc+bias).
// --------------------------------------------------------------------------
template<int ACT, bool RESID>
__global__ void __launch_bounds__(256)
gemm_bf16_wmma(const float* __restrict__ A, int lda,
               const float* __restrict__ W, int ldw,
               const float* __restrict__ bias,
               float* __restrict__ C, int ldc,
               int rows, int K, int M,
               const float* __restrict__ resid,
               const float* __restrict__ scale_ptr)
{
  __shared__ __align__(32) unsigned short Bs[2][16 * 32];   // [buf][col][k] bf16

  const int mT      = M >> 4;
  const int nT      = rows >> 4;
  const int tileCol = blockIdx.x % mT;
  const int rowBlk  = blockIdx.x / mT;
  const int lane    = threadIdx.x & 31;
  const int w       = threadIdx.x >> 5;
  int tileRow = rowBlk * 8 + w;
  if (tileRow >= nT) tileRow = nT - 1;   // clamp: duplicate (identical) work,
                                         // keeps barriers + EXEC uniform
  const int row0 = tileRow << 4;
  const int col0 = tileCol << 4;
  const int kbA  = (lane >> 4) << 4;     // 0 / 16
  const int kbB  = (lane >> 4) << 4;     // 0 / 16
  const float* __restrict__ Arow = A + (size_t)(row0 + (lane & 15)) * lda;

  const int tc = threadIdx.x & 15;       // staging: column
  const int tk = threadIdx.x >> 4;       // staging: k (0..15)

  auto stageB = [&](int k0, int buf) {
    Bs[buf][tc * 32 + tk]      = f2bf(W[(size_t)(k0 + tk)      * ldw + col0 + tc]);
    Bs[buf][tc * 32 + tk + 16] = f2bf(W[(size_t)(k0 + tk + 16) * ldw + col0 + tc]);
  };
  auto loadA = [&](int k0) -> v16bf {
    const float4* a0 = (const float4*)(Arow + k0 + kbA);
    const float4* a1 = (const float4*)(Arow + k0 + 8 + kbA);
    const float4 x0 = a0[0], x1 = a0[1], x2 = a1[0], x3 = a1[1];
    v8u au;
    au[0] = pack2(x0.x, x0.y); au[1] = pack2(x0.z, x0.w);
    au[2] = pack2(x1.x, x1.y); au[3] = pack2(x1.z, x1.w);
    au[4] = pack2(x2.x, x2.y); au[5] = pack2(x2.z, x2.w);
    au[6] = pack2(x3.x, x3.y); au[7] = pack2(x3.z, x3.w);
    return __builtin_bit_cast(v16bf, au);
  };
  auto fragB = [&](int buf) -> v16bf {
    return __builtin_bit_cast(v16bf,
        *(const v8u*)(&Bs[buf][0] + ((lane & 15) << 5) + kbB));
  };

  v8f acc = {};
  if ((K & 63) == 0) {                       // double-buffered: 64 K / barrier pair
    for (int k0 = 0; k0 < K; k0 += 64) {
      __syncthreads();
      stageB(k0, 0);
      stageB(k0 + 32, 1);
      __syncthreads();
      __builtin_prefetch(Arow + k0 + 64, 0, 1);   // speculative, OOB-safe
      acc = __builtin_amdgcn_wmma_f32_16x16x32_bf16(
          false, loadA(k0), false, fragB(0), (short)0, acc, false, false);
      acc = __builtin_amdgcn_wmma_f32_16x16x32_bf16(
          false, loadA(k0 + 32), false, fragB(1), (short)0, acc, false, false);
    }
  } else {                                   // K == 32 cases
    for (int k0 = 0; k0 < K; k0 += 32) {
      __syncthreads();
      stageB(k0, 0);
      __syncthreads();
      acc = __builtin_amdgcn_wmma_f32_16x16x32_bf16(
          false, loadA(k0), false, fragB(0), (short)0, acc, false, false);
    }
  }

  const int   cCol  = col0 + (lane & 15);
  const int   rBase = row0 + ((lane >> 4) << 3);
  const float bv    = bias ? bias[cCol] : 0.f;
  float scale = 0.f;
  if constexpr (RESID) scale = scale_ptr[0];
#pragma unroll
  for (int r = 0; r < 8; ++r) {
    float v = acc[r] + bv;
    if constexpr (ACT == 1) v = fmaxf(v, 0.f);
    if constexpr (ACT == 2) v = __builtin_amdgcn_rcpf(1.f + __expf(-v));
    const size_t idx = (size_t)(rBase + r) * ldc + cCol;
    if constexpr (RESID) v = resid[idx] + scale * v;
    C[idx] = v;
  }
}

// ---------------------------- utility kernels -----------------------------
__global__ void fill_kernel(float* __restrict__ p, long n, float v) {
  long g = (long)blockIdx.x * blockDim.x + threadIdx.x;
  if (g < n) p[g] = v;
}

__global__ void count_kernel(const int* __restrict__ idx, float* __restrict__ cnt, int n) {
  long g = (long)blockIdx.x * blockDim.x + threadIdx.x;
  if (g < n) atomicAdd(&cnt[idx[g]], 1.f);
}

__global__ void scatter_add_rows(const float* __restrict__ vals, int ldv,
                                 const int* __restrict__ idx,
                                 float* __restrict__ out, int ldo,
                                 int C, int nrows) {
  long g = (long)blockIdx.x * blockDim.x + threadIdx.x;
  if (g >= (long)nrows * C) return;
  int r = (int)(g / C), c = (int)(g - (long)r * C);
  atomicAdd(&out[(size_t)idx[r] * ldo + c], vals[(size_t)r * ldv + c]);
}

__global__ void div_by_cnt(float* __restrict__ rows, int ld,
                           const float* __restrict__ cnt, int C, int n) {
  long g = (long)blockIdx.x * blockDim.x + threadIdx.x;
  if (g >= (long)n * C) return;
  int r = (int)(g / C), c = (int)(g - (long)r * C);
  rows[(size_t)r * ld + c] *= __builtin_amdgcn_rcpf(fmaxf(cnt[r], 1.f));
}

__global__ void ew_mul_kernel(float* __restrict__ a, const float* __restrict__ b, long n) {
  long g = (long)blockIdx.x * blockDim.x + threadIdx.x;
  if (g < n) a[g] *= b[g];
}

// ------------------------------ GAT kernels -------------------------------
__global__ void si_sj_kernel(const float* __restrict__ xl, const float* __restrict__ att,
                             int H, int C, int attStride,
                             float* __restrict__ si, float* __restrict__ sj, int n) {
  long g = (long)blockIdx.x * blockDim.x + threadIdx.x;
  if (g >= (long)n * H) return;
  int node = (int)(g / H), h = (int)(g - (long)node * H);
  const float4* xr = (const float4*)(xl + (size_t)node * H * C + (size_t)h * C);
  const float4* ai = (const float4*)(att + (size_t)h * attStride);
  const float4* aj = (const float4*)(att + (size_t)h * attStride + C);
  float s1 = 0.f, s2 = 0.f;
  for (int c = 0; c < (C >> 2); ++c) {
    float4 v = xr[c], a = ai[c], b = aj[c];
    s1 += v.x * a.x + v.y * a.y + v.z * a.z + v.w * a.w;
    s2 += v.x * b.x + v.y * b.y + v.z * b.z + v.w * b.w;
  }
  si[g] = s1; sj[g] = s2;
}

__global__ void alpha_kernel(const float* __restrict__ si, const float* __restrict__ sj,
                             const float* __restrict__ ew, const float* __restrict__ att,
                             int attStride, int aeOff,
                             const int* __restrict__ src, const int* __restrict__ dst,
                             int E_, int total, int H,
                             float* __restrict__ alpha, float* __restrict__ amax) {
  long g = (long)blockIdx.x * blockDim.x + threadIdx.x;
  if (g >= (long)total * H) return;
  int f = (int)(g / H), h = (int)(g - (long)f * H);
  int s = (f < E_) ? src[f] : (f - E_);
  int d = (f < E_) ? dst[f] : (f - E_);
  const float4* ewr = (const float4*)(ew + (size_t)f * 32);
  const float4* ae  = (const float4*)(att + (size_t)h * attStride + aeOff);
  float v = si[(size_t)d * H + h] + sj[(size_t)s * H + h];
#pragma unroll
  for (int j = 0; j < 8; ++j) {
    float4 e = ewr[j], a = ae[j];
    v += e.x * a.x + e.y * a.y + e.z * a.z + e.w * a.w;
  }
  v = (v > 0.f) ? v : 0.2f * v;                // leaky_relu(0.2)
  alpha[g] = v;
  atomicMaxF(&amax[(size_t)d * H + h], v);
}

__global__ void exp_denom_kernel(float* __restrict__ alpha, const float* __restrict__ amax,
                                 float* __restrict__ denom,
                                 const int* __restrict__ dst, int E_, int total, int H) {
  long g = (long)blockIdx.x * blockDim.x + threadIdx.x;
  if (g >= (long)total * H) return;
  int f = (int)(g / H), h = (int)(g - (long)f * H);
  int d = (f < E_) ? dst[f] : (f - E_);
  float e = __expf(alpha[g] - amax[(size_t)d * H + h]);
  alpha[g] = e;
  atomicAdd(&denom[(size_t)d * H + h], e);
}

__global__ void message_kernel(const float* __restrict__ xl, const float* __restrict__ alpha,
                               const float* __restrict__ denom,
                               const int* __restrict__ src, const int* __restrict__ dst,
                               int E_, int total, int H, int C,
                               float* __restrict__ out) {
  long g = (long)blockIdx.x * blockDim.x + threadIdx.x;
  const int HC = H * C;
  if (g >= (long)total * HC) return;
  int f = (int)(g / HC), j = (int)(g - (long)f * HC);
  int h = j / C;
  int s = (f < E_) ? src[f] : (f - E_);
  int d = (f < E_) ? dst[f] : (f - E_);
  float w = alpha[(size_t)f * H + h] *
            __builtin_amdgcn_rcpf(denom[(size_t)d * H + h] + 1e-16f);
  atomicAdd(&out[(size_t)d * HC + j], xl[(size_t)s * HC + j] * w);
}

__global__ void finalize_bn_elu(const float* __restrict__ agg, const float* __restrict__ bias,
                                const float* __restrict__ gg, const float* __restrict__ bb,
                                const float* __restrict__ rm, const float* __restrict__ rv,
                                float* __restrict__ xout, int n, int C) {
  long g = (long)blockIdx.x * blockDim.x + threadIdx.x;
  if (g >= (long)n * C) return;
  int c = (int)(g % C);
  float v = agg[g] + bias[c];
  v = (v - rm[c]) * rsqrtf(rv[c] + 1e-5f) * gg[c] + bb[c];
  xout[g] = (v > 0.f) ? v : (__expf(v) - 1.f);   // elu
}

__global__ void finalize_bias(const float* __restrict__ agg, const float* __restrict__ bias,
                              float* __restrict__ out, int n, int C) {
  long g = (long)blockIdx.x * blockDim.x + threadIdx.x;
  if (g >= (long)n * C) return;
  out[g] = agg[g] + bias[(int)(g % C)];
}

// ==========================================================================
// Host orchestration
// ==========================================================================
extern "C" void kernel_launch(void* const* d_in, const int* in_sizes, int n_in,
                              void* d_out, int out_size, void* d_ws, size_t ws_size,
                              hipStream_t stream)
{
  (void)n_in; (void)out_size; (void)ws_size;
  const int D_IN = 128, E_DIM = 32;
  const int N = in_sizes[0] / D_IN;
  const int E = in_sizes[2] / E_DIM;
  const int T = E + N;                                   // edges + self loops

  const float* x_in = (const float*)d_in[0];
  const int*   src  = (const int*)d_in[1];
  const int*   dstp = src + E;
  const float* eatt = (const float*)d_in[2];
  auto P = [&](int i) { return (const float*)d_in[i]; };
  // Param leaves in setup_inputs() dict insertion order (recursively):
  //  3..10  pre_ad1 {down_w,down_b,up_w,up_b,tp_w,tp_b,fus_w,fus_b}
  // 11..18 post_ad1   19..26 pre_ad2   27..34 post_ad2
  // 35..41 conv1 {lin_w,e1_w,e1_b,e2_w,e2_b,att,bias}
  // 42..48 conv2      49..55 conv3
  // 56..59 bn1 {g,b,rm,rv}   60..63 bn2
  // 64..67 gating {pre1,post1,pre2,post2}

  float* ws = (float*)d_ws;
  size_t off = 0;
  auto alloc = [&](size_t n) { float* p = ws + off; off += n; return p; };
  float* X      = alloc((size_t)N * 256);   // node features (packed, ld = cur channels)
  float* EA     = alloc((size_t)T * 32);    // [edge_attr ; loop_attr]
  float* XL     = alloc((size_t)N * 256);   // per-head linear
  float* AGG    = alloc((size_t)N * 256);   // message accumulation
  size_t szA = (size_t)E * 64; if ((size_t)T * 32 > szA) szA = (size_t)T * 32;
  float* BUFA   = alloc(szA);               // adapter 't' / GAT 'ew'
  float* H1     = alloc((size_t)T * 32);    // edge MLP hidden, padded K 16->32
  float* ALPHA  = alloc((size_t)T * 4);
  float* CONCAT = alloc((size_t)N * 128);   // [h | nt]
  float* FUSED  = alloc((size_t)N * 64);
  float* SI     = alloc((size_t)N * 4);
  float* SJ     = alloc((size_t)N * 4);
  float* AMAX   = alloc((size_t)N * 4);
  float* DEN    = alloc((size_t)N * 4);
  float* CSRC   = alloc((size_t)N);
  float* CDST   = alloc((size_t)N);
  float* PAD    = alloc(32 * 32);           // zero-padded e2_w (16x32 -> 32x32)

  auto cdiv = [](long a, long b) { return (int)((a + b - 1) / b); };
  auto fill = [&](float* p, long n, float v) {
    fill_kernel<<<cdiv(n, 256), 256, 0, stream>>>(p, n, v);
  };
  auto gemm = [&](const float* A, int lda, const float* W, int ldw, const float* bias,
                  float* C, int ldc, int rows, int K, int M, int act,
                  const float* resid, const float* gate) {
    const int mT = M >> 4, nT = rows >> 4;
    const int grid = mT * ((nT + 7) >> 3);       // 8 row-tiles (waves) per block
    if (resid)
      gemm_bf16_wmma<0, true ><<<grid, 256, 0, stream>>>(A, lda, W, ldw, bias, C, ldc, rows, K, M, resid, gate);
    else if (act == 1)
      gemm_bf16_wmma<1, false><<<grid, 256, 0, stream>>>(A, lda, W, ldw, bias, C, ldc, rows, K, M, nullptr, nullptr);
    else if (act == 2)
      gemm_bf16_wmma<2, false><<<grid, 256, 0, stream>>>(A, lda, W, ldw, bias, C, ldc, rows, K, M, nullptr, nullptr);
    else
      gemm_bf16_wmma<0, false><<<grid, 256, 0, stream>>>(A, lda, W, ldw, bias, C, ldc, rows, K, M, nullptr, nullptr);
  };

  // ---- one-time setup: degrees, ea = [edge_attr ; segment_mean(edge_attr,dst)]
  hipMemcpyAsync(X, x_in, (size_t)N * D_IN * sizeof(float), hipMemcpyDeviceToDevice, stream);
  fill(CSRC, N, 0.f);  fill(CDST, N, 0.f);
  count_kernel<<<cdiv(E, 256), 256, 0, stream>>>(src,  CSRC, E);
  count_kernel<<<cdiv(E, 256), 256, 0, stream>>>(dstp, CDST, E);
  hipMemcpyAsync(EA, eatt, (size_t)E * E_DIM * sizeof(float), hipMemcpyDeviceToDevice, stream);
  fill(EA + (size_t)E * E_DIM, (long)N * E_DIM, 0.f);
  scatter_add_rows<<<cdiv((long)E * E_DIM, 256), 256, 0, stream>>>(
      eatt, E_DIM, dstp, EA + (size_t)E * E_DIM, E_DIM, E_DIM, E);
  div_by_cnt<<<cdiv((long)N * E_DIM, 256), 256, 0, stream>>>(
      EA + (size_t)E * E_DIM, E_DIM, CDST, E_DIM, N);

  // ---- adapter: x += g * (relu([relu(x@dw+db) | mean_src(relu(ea@tw+tb))]@fw+fb)@uw + ub)
  auto adapter = [&](int pb, int cin, const float* gate) {
    fill(CONCAT, (long)N * 128, 0.f);
    gemm(X, cin, P(pb + 0), 64, P(pb + 1), CONCAT, 128, N, cin, 64, 1, nullptr, nullptr);
    gemm(eatt, 32, P(pb + 4), 64, P(pb + 5), BUFA, 64, E, 32, 64, 1, nullptr, nullptr);
    scatter_add_rows<<<cdiv((long)E * 64, 256), 256, 0, stream>>>(
        BUFA, 64, src, CONCAT + 64, 128, 64, E);
    div_by_cnt<<<cdiv((long)N * 64, 256), 256, 0, stream>>>(CONCAT + 64, 128, CSRC, 64, N);
    gemm(CONCAT, 128, P(pb + 6), 64, P(pb + 7), FUSED, 64, N, 128, 64, 1, nullptr, nullptr);
    gemm(FUSED, 64, P(pb + 2), cin, P(pb + 3), X, cin, N, 64, cin, 0, X, gate);
  };

  // ---- GAT layer (+ optional bn/elu), result -> outp (packed N x H*C)
  auto gat = [&](int pb, int H, int Cc, int cur, int bnb, float* outp) {
    const int HC = H * Cc;
    const int S  = 2 * Cc + 32;       // att row stride
    gemm(X, cur, P(pb + 0), HC, nullptr, XL, HC, N, cur, HC, 0, nullptr, nullptr);
    // edge MLP: h1 = relu(ea@e1+b1)  (K=32, M=16, stored padded to ld 32)
    fill(H1, (long)T * 32, 0.f);
    gemm(EA, 32, P(pb + 1), 16, P(pb + 2), H1, 32, T, 32, 16, 1, nullptr, nullptr);
    // cw = sigmoid(h1@e2+b2): pad e2_w (16x32) to 32x32 so K%32==0
    fill(PAD, 32 * 32, 0.f);
    hipMemcpyAsync(PAD, P(pb + 3), 16 * 32 * sizeof(float), hipMemcpyDeviceToDevice, stream);
    gemm(H1, 32, PAD, 32, P(pb + 4), BUFA, 32, T, 32, 32, 2, nullptr, nullptr);
    ew_mul_kernel<<<cdiv((long)T * 32, 256), 256, 0, stream>>>(BUFA, EA, (long)T * 32);
    si_sj_kernel<<<cdiv((long)N * H, 256), 256, 0, stream>>>(XL, P(pb + 5), H, Cc, S, SI, SJ, N);
    fill(AMAX, (long)N * H, -1e30f);
    fill(DEN,  (long)N * H, 0.f);
    fill(AGG,  (long)N * HC, 0.f);
    alpha_kernel<<<cdiv((long)T * H, 256), 256, 0, stream>>>(
        SI, SJ, BUFA, P(pb + 5), S, 2 * Cc, src, dstp, E, T, H, ALPHA, AMAX);
    exp_denom_kernel<<<cdiv((long)T * H, 256), 256, 0, stream>>>(ALPHA, AMAX, DEN, dstp, E, T, H);
    message_kernel<<<cdiv((long)T * HC, 256), 256, 0, stream>>>(
        XL, ALPHA, DEN, src, dstp, E, T, H, Cc, AGG);
    if (bnb >= 0)
      finalize_bn_elu<<<cdiv((long)N * HC, 256), 256, 0, stream>>>(
          AGG, P(pb + 6), P(bnb + 0), P(bnb + 1), P(bnb + 2), P(bnb + 3), outp, N, HC);
    else
      finalize_bias<<<cdiv((long)N * HC, 256), 256, 0, stream>>>(AGG, P(pb + 6), outp, N, HC);
  };

  // ---- forward sequence ----
  adapter(3, 128, P(64));            // pre_ad1, gate pre1
  gat(35, 4, 64, 128, 56, X);        // conv1 -> X (256 ch), bn1 + elu
  adapter(11, 256, P(65));           // post_ad1
  adapter(19, 256, P(66));           // pre_ad2
  gat(42, 4, 32, 256, 60, X);        // conv2 -> X (128 ch), bn2 + elu
  adapter(27, 128, P(67));           // post_ad2
  gat(49, 1, 64, 128, -1, (float*)d_out);  // conv3 -> output (N x 64)
}
